// LSNNRecurrent_88098369175817
// MI455X (gfx1250) — matrix operand order repaired and checked by reading, not compiled, not run on hardware
//
#include <hip/hip_runtime.h>

// ---------------------------------------------------------------------------
// LSNN recurrent scan for MI455X (gfx1250, wave32, WMMA).
//
//  * Input GEMM cur = x @ w_in^T has no cross-step dependency: computed as one
//    big parallel bf16x3 GEMM (memory bound, ~192MB traffic ~ 8us at 23.3TB/s)
//    when ws_size permits the 128MB scratch; otherwise fused fallback.
//  * Serial scan: one launch per step; recurrent GEMM z@w_rec^T on
//    v_wmma_f32_16x16x32_bf16 with hi/lo bf16 weight split (z is exactly
//    {0,1} in bf16 -> near-fp32 accuracy). FOUR independent accumulators
//    (K halves x hi/lo) keep the WMMA RAW chain short on the serial path.
//  * Weights decomposed to bf16 hi/lo once; resident in 192MB L2.
// ---------------------------------------------------------------------------

typedef __attribute__((ext_vector_type(16))) __bf16 v16bf;
typedef __attribute__((ext_vector_type(8)))  __bf16 v8bf;
typedef __attribute__((ext_vector_type(8)))  float  v8f;
typedef __attribute__((ext_vector_type(4)))  float  v4f;

#define LSNN_T    512
#define LSNN_B    64
#define LSNN_NIN  512
#define LSNN_H    1024
#define LSNN_BH   (LSNN_B * LSNN_H)          // 65536
#define LSNN_M    (LSNN_T * LSNN_B)          // 32768 rows of the input GEMM

union BF16x16 { v16bf v; v8bf h[2]; };

// A-matrix fragment (16x32 bf16, M x K), ISA 16-bit A layout:
// lane l: row M = l&15; VGPR0-3 hold K = (l>>4)*8 + 0..7, VGPR4-7 hold K+16.
static __device__ inline v16bf load_a_bf16(const __bf16* __restrict__ base,
                                           int ld, int row, int k0, int lane) {
  const int khalf = (lane >> 4) * 8;
  const __bf16* p = base + (size_t)row * ld + k0 + khalf;
  BF16x16 u;
  u.h[0] = *(const v8bf*)(p);
  u.h[1] = *(const v8bf*)(p + 16);
  return u.v;
}

// B-matrix fragment (32x16 bf16, K x N), ISA 16-bit B layout:
// lane l: col N = l&15; VGPR0-3 hold K = k0 + (l>>4)*8 + 0..7, VGPR4-7 hold K+16.
// Source is w[N_global][K] row-major (w^T as the B operand).
static __device__ inline v16bf load_b_bf16(const __bf16* __restrict__ w,
                                           int ld, int hcol, int k0, int lane) {
  const __bf16* p = w + (size_t)hcol * ld + k0 + ((lane >> 4) * 8);
  BF16x16 u;
  u.h[0] = *(const v8bf*)(p);
  u.h[1] = *(const v8bf*)(p + 16);
  return u.v;
}

// A fragment from f32 source with on-the-fly hi/lo bf16 decomposition.
static __device__ inline void load_a_f32_hilo(const float* __restrict__ base,
                                              int ld, int row, int k0, int lane,
                                              v16bf& hi, v16bf& lo) {
  const int khalf = (lane >> 4) * 8;
  const float* p = base + (size_t)row * ld + k0 + khalf;
  float f[16];
  *(v4f*)(f + 0)  = *(const v4f*)(p + 0);
  *(v4f*)(f + 4)  = *(const v4f*)(p + 4);
  *(v4f*)(f + 8)  = *(const v4f*)(p + 16);
  *(v4f*)(f + 12) = *(const v4f*)(p + 20);
#pragma unroll
  for (int j = 0; j < 16; ++j) {
    __bf16 h = (__bf16)f[j];
    hi[j] = h;
    lo[j] = (__bf16)(f[j] - (float)h);
  }
}

#define WMMA_BF16(A, Bm, C) \
  __builtin_amdgcn_wmma_f32_16x16x32_bf16(false, (A), false, (Bm), (short)0, (C), false, false)

// --------------------------- prep kernels ----------------------------------

__global__ void lsnn_cvt_hilo_kernel(const float* __restrict__ w,
                                     __bf16* __restrict__ wh,
                                     __bf16* __restrict__ wl, int n) {
  int i = blockIdx.x * blockDim.x + threadIdx.x;
  if (i < n) {
    float f = w[i];
    __bf16 h = (__bf16)f;
    wh[i] = h;
    wl[i] = (__bf16)(f - (float)h);
  }
}

__global__ void lsnn_init_state_kernel(float* __restrict__ v,
                                       float* __restrict__ cur,
                                       float* __restrict__ badp,
                                       __bf16* __restrict__ z0,
                                       __bf16* __restrict__ z1, int n) {
  int i = blockIdx.x * blockDim.x + threadIdx.x;
  if (i < n) {
    v[i]    = 0.0f;   // V_LEAK
    cur[i]  = 0.0f;
    badp[i] = 1.0f;   // VTH
    z0[i]   = (__bf16)0.0f;
    z1[i]   = (__bf16)0.0f;
  }
}

// ----------------------- big input GEMM (parallel) --------------------------
// curbuf[M=32768, H=1024] = x[M, 512] @ w_in^T, bf16x3 split.
// One wave -> 1 M-tile x 4 consecutive N-tiles (A fragment + f32->bf16
// conversion amortized 4x; 4 independent accumulator chains).
// Grid: 4096 blocks x 256 threads = 32768 waves = 2048 Mtiles x 16 Ngroups.

__global__ __launch_bounds__(256) void lsnn_ingemm_kernel(
    const float*  __restrict__ x,      // [M, NIN]
    const __bf16* __restrict__ whin,   // [H, NIN]
    const __bf16* __restrict__ wlin,   // [H, NIN]
    float* __restrict__ curbuf) {      // [M, H]
  const int lane  = threadIdx.x & 31;
  const int wave  = blockIdx.x * (blockDim.x >> 5) + (threadIdx.x >> 5);
  const int mtile = wave >> 4;          // 0..2047
  const int ngrp  = wave & 15;          // 0..15 -> N tiles ngrp*4 .. +3
  const int arow  = mtile * 16 + (lane & 15);
  const int ncol  = lane & 15;

  v8f acc[4] = {};
  for (int k0 = 0; k0 < LSNN_NIN; k0 += 32) {
    v16bf ah, al;
    load_a_f32_hilo(x, LSNN_NIN, arow, k0, lane, ah, al);
#pragma unroll
    for (int j = 0; j < 4; ++j) {
      const int hcol = (ngrp * 4 + j) * 16 + ncol;
      v16bf bh = load_b_bf16(whin, LSNN_NIN, hcol, k0, lane);
      v16bf bl = load_b_bf16(wlin, LSNN_NIN, hcol, k0, lane);
      acc[j] = WMMA_BF16(ah, bh, acc[j]);
      acc[j] = WMMA_BF16(ah, bl, acc[j]);
      acc[j] = WMMA_BF16(al, bh, acc[j]);
    }
  }
  const int mhalf = (lane >> 4) * 8;
#pragma unroll
  for (int j = 0; j < 4; ++j) {
    const int hcol = (ngrp * 4 + j) * 16 + ncol;
#pragma unroll
    for (int r = 0; r < 8; ++r) {
      curbuf[(size_t)(mtile * 16 + mhalf + r) * LSNN_H + hcol] = acc[j][r];
    }
  }
}

// -------------------- step kernel: recurrent only (path A) ------------------
// Grid: 32 blocks x 256 threads = 256 wave32s; one 16x16 tile per wave.
// K=1024 split into two halves x (hi/lo) -> 4 independent WMMA chains of 16.

__global__ __launch_bounds__(256) void lsnn_step_rec_kernel(
    const float*  __restrict__ curbuf,  // [M, H] precomputed input currents
    const __bf16* __restrict__ whrec,   // [H, H]
    const __bf16* __restrict__ wlrec,   // [H, H]
    float* __restrict__ v,
    float* __restrict__ cur,
    float* __restrict__ badp,
    const __bf16* __restrict__ zin,
    __bf16* __restrict__ zout,
    float* __restrict__ out,
    int t) {
  const int lane = threadIdx.x & 31;
  const int wave = blockIdx.x * (blockDim.x >> 5) + (threadIdx.x >> 5);
  const int mt = wave & 3;
  const int nt = wave >> 2;
  const int arow = mt * 16 + (lane & 15);
  const int hcol = nt * 16 + (lane & 15);

  v8f a0 = {}, a1 = {}, a2 = {}, a3 = {};
  for (int k0 = 0; k0 < LSNN_H / 2; k0 += 32) {
    const int k1 = k0 + LSNN_H / 2;
    v16bf za  = load_a_bf16(zin, LSNN_H, arow, k0, lane);
    v16bf bh  = load_b_bf16(whrec, LSNN_H, hcol, k0, lane);
    v16bf bl  = load_b_bf16(wlrec, LSNN_H, hcol, k0, lane);
    v16bf zb  = load_a_bf16(zin, LSNN_H, arow, k1, lane);
    v16bf bh2 = load_b_bf16(whrec, LSNN_H, hcol, k1, lane);
    v16bf bl2 = load_b_bf16(wlrec, LSNN_H, hcol, k1, lane);
    a0 = WMMA_BF16(za, bh,  a0);
    a1 = WMMA_BF16(za, bl,  a1);
    a2 = WMMA_BF16(zb, bh2, a2);
    a3 = WMMA_BF16(zb, bl2, a3);
  }
  const v8f acc = (a0 + a2) + (a1 + a3);

  const float* curt = curbuf + (size_t)t * LSNN_BH;
  const int mhalf = (lane >> 4) * 8;
#pragma unroll
  for (int r = 0; r < 8; ++r) {
    const int brow = mt * 16 + mhalf + r;
    const size_t idx = (size_t)brow * LSNN_H + hcol;
    const float vv = v[idx];
    const float ii = cur[idx];
    const float bb = badp[idx];
    const float v_dec = vv + 0.1f * (ii - vv);
    const float i_dec = 0.8f * ii;
    const float b_dec = bb + 1.25e-6f * (1.0f - bb);
    const float z = (v_dec - b_dec) > 0.0f ? 1.0f : 0.0f;
    const float v_new = (1.0f - z) * v_dec;
    const float i_new = i_dec + acc[r] + curt[idx];
    const float b_new = b_dec + z * 1.8f;
    v[idx]    = v_new;
    cur[idx]  = i_new;
    badp[idx] = b_new;
    zout[idx] = (__bf16)z;
    out[(size_t)t * LSNN_BH + idx] = z;
    if (t == LSNN_T - 1) {
      const size_t F = (size_t)LSNN_T * LSNN_BH;
      out[F + 0 * (size_t)LSNN_BH + idx] = z;
      out[F + 1 * (size_t)LSNN_BH + idx] = v_new;
      out[F + 2 * (size_t)LSNN_BH + idx] = i_new;
      out[F + 3 * (size_t)LSNN_BH + idx] = b_new;
    }
  }
}

// ----------------- step kernel: fused fallback (path B) ---------------------

__global__ __launch_bounds__(256) void lsnn_step_fused_kernel(
    const float*  __restrict__ x,
    const __bf16* __restrict__ whin,
    const __bf16* __restrict__ wlin,
    const __bf16* __restrict__ whrec,
    const __bf16* __restrict__ wlrec,
    float* __restrict__ v,
    float* __restrict__ cur,
    float* __restrict__ badp,
    const __bf16* __restrict__ zin,
    __bf16* __restrict__ zout,
    float* __restrict__ out,
    int t) {
  const int lane = threadIdx.x & 31;
  const int wave = blockIdx.x * (blockDim.x >> 5) + (threadIdx.x >> 5);
  const int mt = wave & 3;
  const int nt = wave >> 2;
  const int arow = mt * 16 + (lane & 15);
  const int hcol = nt * 16 + (lane & 15);

  v8f a0 = {}, a1 = {}, a2 = {}, a3 = {};
  for (int k0 = 0; k0 < LSNN_H / 2; k0 += 32) {
    const int k1 = k0 + LSNN_H / 2;
    v16bf za  = load_a_bf16(zin, LSNN_H, arow, k0, lane);
    v16bf bh  = load_b_bf16(whrec, LSNN_H, hcol, k0, lane);
    v16bf bl  = load_b_bf16(wlrec, LSNN_H, hcol, k0, lane);
    v16bf zb  = load_a_bf16(zin, LSNN_H, arow, k1, lane);
    v16bf bh2 = load_b_bf16(whrec, LSNN_H, hcol, k1, lane);
    v16bf bl2 = load_b_bf16(wlrec, LSNN_H, hcol, k1, lane);
    a0 = WMMA_BF16(za, bh,  a0);
    a1 = WMMA_BF16(za, bl,  a1);
    a2 = WMMA_BF16(zb, bh2, a2);
    a3 = WMMA_BF16(zb, bl2, a3);
  }
  const float* xt = x + (size_t)t * (LSNN_B * LSNN_NIN);
  for (int k0 = 0; k0 < LSNN_NIN; k0 += 32) {
    v16bf ah, al;
    load_a_f32_hilo(xt, LSNN_NIN, arow, k0, lane, ah, al);
    v16bf bh = load_b_bf16(whin, LSNN_NIN, hcol, k0, lane);
    v16bf bl = load_b_bf16(wlin, LSNN_NIN, hcol, k0, lane);
    a0 = WMMA_BF16(ah, bh, a0);
    a1 = WMMA_BF16(ah, bl, a1);
    a2 = WMMA_BF16(al, bh, a2);
  }
  const v8f acc = (a0 + a2) + (a1 + a3);

  const int mhalf = (lane >> 4) * 8;
#pragma unroll
  for (int r = 0; r < 8; ++r) {
    const int brow = mt * 16 + mhalf + r;
    const size_t idx = (size_t)brow * LSNN_H + hcol;
    const float vv = v[idx];
    const float ii = cur[idx];
    const float bb = badp[idx];
    const float v_dec = vv + 0.1f * (ii - vv);
    const float i_dec = 0.8f * ii;
    const float b_dec = bb + 1.25e-6f * (1.0f - bb);
    const float z = (v_dec - b_dec) > 0.0f ? 1.0f : 0.0f;
    const float v_new = (1.0f - z) * v_dec;
    const float i_new = i_dec + acc[r];
    const float b_new = b_dec + z * 1.8f;
    v[idx]    = v_new;
    cur[idx]  = i_new;
    badp[idx] = b_new;
    zout[idx] = (__bf16)z;
    out[(size_t)t * LSNN_BH + idx] = z;
    if (t == LSNN_T - 1) {
      const size_t F = (size_t)LSNN_T * LSNN_BH;
      out[F + 0 * (size_t)LSNN_BH + idx] = z;
      out[F + 1 * (size_t)LSNN_BH + idx] = v_new;
      out[F + 2 * (size_t)LSNN_BH + idx] = i_new;
      out[F + 3 * (size_t)LSNN_BH + idx] = b_new;
    }
  }
}

// ----------------------------- launcher ------------------------------------

extern "C" void kernel_launch(void* const* d_in, const int* in_sizes, int n_in,
                              void* d_out, int out_size, void* d_ws, size_t ws_size,
                              hipStream_t stream) {
  const float* x     = (const float*)d_in[0];   // [512, 64, 512]
  const float* w_in  = (const float*)d_in[1];   // [1024, 512]
  const float* w_rec = (const float*)d_in[2];   // [1024, 1024]
  float* out = (float*)d_out;

  char* ws = (char*)d_ws;
  size_t off = 0;
  __bf16* whrec = (__bf16*)(ws + off); off += (size_t)LSNN_H * LSNN_H * 2;
  __bf16* wlrec = (__bf16*)(ws + off); off += (size_t)LSNN_H * LSNN_H * 2;
  __bf16* whin  = (__bf16*)(ws + off); off += (size_t)LSNN_H * LSNN_NIN * 2;
  __bf16* wlin  = (__bf16*)(ws + off); off += (size_t)LSNN_H * LSNN_NIN * 2;
  float*  v     = (float*)(ws + off);  off += (size_t)LSNN_BH * 4;
  float*  cur   = (float*)(ws + off);  off += (size_t)LSNN_BH * 4;
  float*  badp  = (float*)(ws + off);  off += (size_t)LSNN_BH * 4;
  __bf16* z0    = (__bf16*)(ws + off); off += (size_t)LSNN_BH * 2;
  __bf16* z1    = (__bf16*)(ws + off); off += (size_t)LSNN_BH * 2;
  float*  curbuf = (float*)(ws + off);
  const size_t need_big = off + (size_t)LSNN_M * LSNN_H * 4;   // +128MB
  const bool big = (ws_size >= need_big);      // constant per process -> deterministic

  {
    int n = LSNN_H * LSNN_H;
    lsnn_cvt_hilo_kernel<<<(n + 255) / 256, 256, 0, stream>>>(w_rec, whrec, wlrec, n);
  }
  {
    int n = LSNN_H * LSNN_NIN;
    lsnn_cvt_hilo_kernel<<<(n + 255) / 256, 256, 0, stream>>>(w_in, whin, wlin, n);
  }
  lsnn_init_state_kernel<<<(LSNN_BH + 255) / 256, 256, 0, stream>>>(v, cur, badp, z0, z1, LSNN_BH);

  if (big) {
    // Fully parallel input GEMM off the serial critical path.
    lsnn_ingemm_kernel<<<4096, 256, 0, stream>>>(x, whin, wlin, curbuf);
    for (int t = 0; t < LSNN_T; ++t) {
      const __bf16* zi = (t & 1) ? z1 : z0;
      __bf16*       zo = (t & 1) ? z0 : z1;
      lsnn_step_rec_kernel<<<32, 256, 0, stream>>>(curbuf, whrec, wlrec,
                                                   v, cur, badp, zi, zo, out, t);
    }
  } else {
    for (int t = 0; t < LSNN_T; ++t) {
      const __bf16* zi = (t & 1) ? z1 : z0;
      __bf16*       zo = (t & 1) ? z0 : z1;
      lsnn_step_fused_kernel<<<32, 256, 0, stream>>>(x, whin, wlin, whrec, wlrec,
                                                     v, cur, badp, zi, zo, out, t);
    }
  }
}
